// ModalCoTReasoning_88536455840319
// MI455X (gfx1250) — matrix operand order repaired and checked by reading, not compiled, not run on hardware
//
#include <hip/hip_runtime.h>
#include <math.h>

// ---------------------------------------------------------------------------
// ModalCoTReasoning on MI455X (gfx1250):
//  * all contractions via v_wmma_f32_16x16x32_bf16 (wave32, f32 accumulate)
//  * A/B/V tiles staged LDS via the Tensor Data Mover (tensor_load_to_lds,
//    TENSORcnt, s_wait_tensorcnt), double-buffered so DMA overlaps WMMA
//  * fused attention per 16-query tile; probabilities kept as bf16 in LDS
// ---------------------------------------------------------------------------

typedef __attribute__((ext_vector_type(16))) __bf16 bf16x16;
typedef __attribute__((ext_vector_type(8)))  float  f32x8;
typedef __attribute__((ext_vector_type(4)))  unsigned int u32x4;
typedef __attribute__((ext_vector_type(8)))  int i32x8;
typedef __attribute__((ext_vector_type(4)))  int i32x4;

#define C_DIM   768
#define T_SEQ   1024
#define B_SZ    8
#define H_NUM   8
#define D_HEAD  96
#define M_TOK   (B_SZ * T_SEQ)          // 8192 tokens
#define NEL     ((size_t)M_TOK * C_DIM) // 6291456 elements

#if defined(__has_builtin)
#if __has_builtin(__builtin_amdgcn_tensor_load_to_lds)
#define HAVE_TDM 1
#endif
#if __has_builtin(__builtin_amdgcn_s_wait_tensorcnt)
#define HAVE_TCNT 1
#endif
#endif

#if defined(HAVE_TDM) && __has_include(<hip/amd_detail/amd_gfx1250_TDM.h>)
#define TDM_SIX_ARG 1
#endif

#if defined(HAVE_TDM) && defined(HAVE_TCNT)
#define TDM_WAIT(n) __builtin_amdgcn_s_wait_tensorcnt(n)
#else
#define TDM_WAIT(n) ((void)0)
#endif

// Issue one 2D TDM tile load: rows x width f32 elements, row stride `stride`
// (elements) in global, packed contiguously at lds_dst. Call from ONE wave;
// that wave later does TDM_WAIT, everyone else syncs via __syncthreads().
__device__ __forceinline__ void tdm_load_tile(float* lds_dst, const float* gsrc,
                                              unsigned width, unsigned rows,
                                              unsigned long long stride_elems) {
#if defined(HAVE_TDM)
  const unsigned lofs = (unsigned)(unsigned long long)(uintptr_t)lds_dst; // LDS byte offset
  const unsigned long long ga = (unsigned long long)(uintptr_t)gsrc;
  u32x4 g0;
  g0[0] = 1u;                                             // count=1, user descriptor
  g0[1] = lofs;                                           // lds_addr (bytes)
  g0[2] = (unsigned)ga;                                   // global_addr[31:0]
  g0[3] = (unsigned)((ga >> 32) & 0x01FFFFFFu) | (2u << 30); // addr[56:32] | type=2
  i32x8 g1;
  g1[0] = (int)(2u << 16);                                // data_size = 4 bytes
  g1[1] = (int)((width & 0xFFFFu) << 16);                 // tensor_dim0[15:0]
  g1[2] = (int)(((width >> 16) & 0xFFFFu) | ((rows & 0xFFFFu) << 16)); // td0_hi|td1_lo
  g1[3] = (int)(((rows >> 16) & 0xFFFFu) | ((width & 0xFFFFu) << 16)); // td1_hi|tile_dim0
  g1[4] = (int)(rows & 0xFFFFu);                          // tile_dim1 (tile_dim2 = 0)
  g1[5] = (int)(unsigned)(stride_elems & 0xFFFFFFFFull);  // dim0_stride[31:0]
  g1[6] = (int)(unsigned)((stride_elems >> 32) & 0xFFFFull); // dim0_stride[47:32]
  g1[7] = 0;
  i32x4 z4 = {0, 0, 0, 0};
#if defined(TDM_SIX_ARG)
  i32x8 z8 = {0, 0, 0, 0, 0, 0, 0, 0};
  __builtin_amdgcn_tensor_load_to_lds(g0, g1, z4, z4, z8, 0);
#else
  __builtin_amdgcn_tensor_load_to_lds(g0, g1, z4, z4, 0);
#endif
#else
  // fallback: the issuing wave copies the tile cooperatively
  const int l = threadIdx.x & 31;
  for (unsigned idx = (unsigned)l; idx < width * rows; idx += 32) {
    unsigned r = idx / width, c = idx % width;
    lds_dst[idx] = gsrc[(size_t)r * stride_elems + c];
  }
#endif
}

__device__ __forceinline__ float warp_max32(float v) {
#pragma unroll
  for (int off = 16; off > 0; off >>= 1) v = fmaxf(v, __shfl_xor(v, off, 32));
  return v;
}
__device__ __forceinline__ float warp_sum32(float v) {
#pragma unroll
  for (int off = 16; off > 0; off >>= 1) v += __shfl_xor(v, off, 32);
  return v;
}

// ---------------------------------------------------------------------------
// Folded Kripke mixer weights: cur = x @ Weff^T + beff
// ---------------------------------------------------------------------------
__global__ void prep_weff_kernel(const float* __restrict__ mixer_W,
                                 const float* __restrict__ acc,
                                 float* __restrict__ Weff) {
  int idx = blockIdx.x * blockDim.x + threadIdx.x;
  if (idx >= C_DIM * C_DIM) return;
  int o = idx / C_DIM, c = idx % C_DIM;
  float s = 0.f;
#pragma unroll
  for (int i = 0; i < 4; ++i) {
    float si = acc[i * 4 + 0] + acc[i * 4 + 1] + acc[i * 4 + 2] + acc[i * 4 + 3];
    s += si * mixer_W[(size_t)o * (4 * C_DIM) + i * C_DIM + c];
  }
  Weff[idx] = s;
}

__global__ void prep_beff_kernel(const float* __restrict__ mixer_W,
                                 const float* __restrict__ mixer_b,
                                 const float* __restrict__ acc,
                                 const float* __restrict__ world_emb,
                                 float* __restrict__ beff) {
  int o = blockIdx.x * blockDim.x + threadIdx.x;
  if (o >= C_DIM) return;
  float s = mixer_b[o];
  for (int i = 0; i < 4; ++i) {
    const float* mw = mixer_W + (size_t)o * (4 * C_DIM) + i * C_DIM;
    float a0 = acc[i * 4 + 0], a1 = acc[i * 4 + 1], a2 = acc[i * 4 + 2], a3 = acc[i * 4 + 3];
    for (int c = 0; c < C_DIM; ++c) {
      float w = a0 * world_emb[0 * C_DIM + c] + a1 * world_emb[1 * C_DIM + c] +
                a2 * world_emb[2 * C_DIM + c] + a3 * world_emb[3 * C_DIM + c];
      s += mw[c] * w;
    }
  }
  beff[o] = s;
}

// ---------------------------------------------------------------------------
// Y[M,N] = X[M,K] @ W[N,K]^T + bias (+act). 8 waves / block, 64x32 block tile,
// 16x16 per-wave WMMA tile. A (64x32) and B (32x32) k-slabs staged in LDS by
// the TDM, double-buffered (wave 0 issues, waits TENSORcnt, block barriers).
// A/B fragments use contiguous k order, k = 16*(lane/16) + s (wave32).
// ---------------------------------------------------------------------------
__global__ __launch_bounds__(256)
void linear_wmma_kernel(const float* __restrict__ X, const float* __restrict__ W,
                        const float* __restrict__ bias, float* __restrict__ Y,
                        int M, int N, int K, int act) {
  __shared__ float At[2][64 * 32];   // 16 KB
  __shared__ float Bt[2][32 * 32];   //  8 KB

  const int lane = threadIdx.x & 31;
  const int wave = threadIdx.x >> 5;
  const int mblk = blockIdx.x * 64;
  const int nblk = blockIdx.y * 32;
  const int rowl = (wave & 3) * 16 + (lane & 15);   // 0..63
  const int coll = (wave >> 2) * 16 + (lane & 15);  // 0..31
  const int khalfA = (lane < 16) ? 0 : 16;
  const int kbaseB = (lane < 16) ? 0 : 16;

  if (wave == 0) {
    tdm_load_tile(At[0], X + (size_t)mblk * K, 32, 64, (unsigned long long)K);
    tdm_load_tile(Bt[0], W + (size_t)nblk * K, 32, 32, (unsigned long long)K);
  }

  f32x8 acc = {};
  int buf = 0;
  for (int kk = 0; kk < K; kk += 32) {
    if (wave == 0) {
      if (kk + 32 < K) {
        tdm_load_tile(At[buf ^ 1], X + (size_t)mblk * K + kk + 32, 32, 64, (unsigned long long)K);
        tdm_load_tile(Bt[buf ^ 1], W + (size_t)nblk * K + kk + 32, 32, 32, (unsigned long long)K);
        TDM_WAIT(2);   // current slab complete; next slab still in flight
      } else {
        TDM_WAIT(0);
      }
    }
    __syncthreads();

    bf16x16 a, b;
    const float* ap = &At[buf][rowl * 32 + khalfA];
    const float* bp = &Bt[buf][coll * 32 + kbaseB];
#pragma unroll
    for (int i = 0; i < 8; ++i) {
      a[i]     = (__bf16)ap[i];
      a[8 + i] = (__bf16)ap[8 + i];
    }
#pragma unroll
    for (int i = 0; i < 16; ++i) b[i] = (__bf16)bp[i];
    acc = __builtin_amdgcn_wmma_f32_16x16x32_bf16(false, a, false, b,
                                                  (short)0, acc, false, false);
    __syncthreads();
    buf ^= 1;
  }

  const int col = nblk + coll;
  const float bv = bias ? bias[col] : 0.0f;
#pragma unroll
  for (int v = 0; v < 8; ++v) {
    int m = mblk + (wave & 3) * 16 + v + ((lane >> 4) << 3);
    float r = acc[v] + bv;
    if (act == 1) r = fmaxf(r, 0.0f);                  // relu
    else if (act == 2) r = 1.0f / (1.0f + __expf(-r)); // sigmoid
    Y[(size_t)m * N + col] = r;
  }
}

// ---------------------------------------------------------------------------
// Fused modal attention, one 16-query tile per block (4 cooperating waves).
//   scores -> bf16 probs in LDS -> O = P V with TDM-staged V k-slabs.
// The first V slab is issued before the score phase so the DMA hides there.
// ---------------------------------------------------------------------------
__global__ __launch_bounds__(128)
void attn_wmma_kernel(const float* __restrict__ Q, const float* __restrict__ K,
                      const float* __restrict__ V, const float* __restrict__ acc4,
                      const float* __restrict__ modal_w_p, int use_necessity,
                      float* __restrict__ O) {
  __shared__ __bf16 S[16 * T_SEQ];    // 32 KB probability row-block
  __shared__ float  Vt[2][32 * D_HEAD]; // 24 KB double-buffered V slab

  const int lane = threadIdx.x & 31;
  const int wave = threadIdx.x >> 5;          // 0..3
  const int bh = blockIdx.x;                  // b*H + h
  const int b  = bh / H_NUM, h = bh % H_NUM;
  const int q0 = blockIdx.y * 16;

  const float modal_w = modal_w_p[0];
  const float scale = rsqrtf((float)D_HEAD);
  const int khalfA = (lane < 16) ? 0 : 16;
  const int kbaseB = (lane < 16) ? 0 : 16;

  // kick off the first V slab now; it completes under the score/softmax phases
  if (wave == 0)
    tdm_load_tile(Vt[0], V + (size_t)(b * T_SEQ) * C_DIM + h * D_HEAD,
                  D_HEAD, 32, (unsigned long long)C_DIM);

  // ---- preload Q fragments (16 x 96 = 3 k-steps), scale folded in ----
  bf16x16 qf[3];
  const int qrow = q0 + (lane & 15);
  const size_t qbase = ((size_t)(b * T_SEQ + qrow)) * C_DIM + h * D_HEAD;
#pragma unroll
  for (int s = 0; s < 3; ++s) {
    const float* qp = Q + qbase + s * 32 + khalfA;
#pragma unroll
    for (int i = 0; i < 8; ++i) {
      qf[s][i]     = (__bf16)(qp[i] * scale);
      qf[s][8 + i] = (__bf16)(qp[8 + i] * scale);
    }
  }

  // ---- phase 1: scores, kt tiles strided across the 4 waves ----
  for (int kt = wave; kt < T_SEQ / 16; kt += 4) {
    f32x8 c = {};
    const int key = kt * 16 + (lane & 15);
    const size_t kbase = ((size_t)(b * T_SEQ + key)) * C_DIM + h * D_HEAD;
#pragma unroll
    for (int s = 0; s < 3; ++s) {
      bf16x16 bf;
      const float* kp = K + kbase + s * 32 + kbaseB;
#pragma unroll
      for (int i = 0; i < 16; ++i) bf[i] = (__bf16)kp[i];
      c = __builtin_amdgcn_wmma_f32_16x16x32_bf16(false, qf[s], false, bf,
                                                  (short)0, c, false, false);
    }
#pragma unroll
    for (int v = 0; v < 8; ++v) {
      const int qm = v + ((lane >> 4) << 3);
      const int qg = q0 + qm;
      const int kg = kt * 16 + (lane & 15);
      const float mask = acc4[(qg & 3) * 4 + (kg & 3)];   // acc[t%4, t'%4], W==4
      float val = c[v];
      if (use_necessity) { if (mask == 0.0f) val = -__builtin_inff(); }
      else               { val += modal_w * mask; }
      S[qm * T_SEQ + kg] = (__bf16)val;
    }
  }
  __syncthreads();

  // ---- phase 2: softmax, 4 rows per wave (f32 math over bf16 storage) ----
  for (int r = wave * 4; r < wave * 4 + 4; ++r) {
    __bf16* row = &S[r * T_SEQ];
    float mx = -__builtin_inff();
    for (int c = lane; c < T_SEQ; c += 32) mx = fmaxf(mx, (float)row[c]);
    mx = warp_max32(mx);
    float sum = 0.f;
    for (int c = lane; c < T_SEQ; c += 32) {
      float e = __expf((float)row[c] - mx);
      row[c] = (__bf16)e;
      sum += e;
    }
    sum = warp_sum32(sum);
    const float inv = 1.0f / sum;
    for (int c = lane; c < T_SEQ; c += 32) row[c] = (__bf16)((float)row[c] * inv);
  }
  __syncthreads();

  // ---- phase 3: O = P V; V slabs double-buffered through the TDM ----
  f32x8 oacc[2] = {{}, {}};
  int buf = 0;
  for (int kk = 0; kk < T_SEQ; kk += 32) {
    if (wave == 0) {
      if (kk + 32 < T_SEQ) {
        tdm_load_tile(Vt[buf ^ 1],
                      V + (size_t)(b * T_SEQ + kk + 32) * C_DIM + h * D_HEAD,
                      D_HEAD, 32, (unsigned long long)C_DIM);
        TDM_WAIT(1);
      } else {
        TDM_WAIT(0);
      }
    }
    __syncthreads();

#pragma unroll
    for (int t = 0; t < 2; ++t) {
      const int nt = wave + t * 4;          // wave-uniform predicate
      if (nt < D_HEAD / 16) {
        bf16x16 a, bf;
        const __bf16* sp = &S[(lane & 15) * T_SEQ + kk + khalfA];
#pragma unroll
        for (int i = 0; i < 8; ++i) {
          a[i]     = sp[i];
          a[8 + i] = sp[8 + i];
        }
        const int vcol = nt * 16 + (lane & 15);
#pragma unroll
        for (int i = 0; i < 16; ++i) bf[i] = (__bf16)Vt[buf][(kbaseB + i) * D_HEAD + vcol];
        oacc[t] = __builtin_amdgcn_wmma_f32_16x16x32_bf16(false, a, false, bf,
                                                          (short)0, oacc[t], false, false);
      }
    }
    __syncthreads();
    buf ^= 1;
  }

#pragma unroll
  for (int t = 0; t < 2; ++t) {
    const int nt = wave + t * 4;
    if (nt < D_HEAD / 16) {
#pragma unroll
      for (int v = 0; v < 8; ++v) {
        const int qm = v + ((lane >> 4) << 3);
        O[((size_t)(b * T_SEQ + q0 + qm)) * C_DIM + h * D_HEAD + nt * 16 + (lane & 15)] =
            oacc[t][v];
      }
    }
  }
}

// ---------------------------------------------------------------------------
// Early-exit machinery (graph-capture safe, no host readback).
// ---------------------------------------------------------------------------
__global__ void reduce_sum_kernel(const float* __restrict__ x, long long n,
                                  float* __restrict__ out) {
  float s = 0.f;
  for (long long i = blockIdx.x * (long long)blockDim.x + threadIdx.x; i < n;
       i += (long long)gridDim.x * blockDim.x)
    s += x[i];
  s = warp_sum32(s);
  if ((threadIdx.x & 31) == 0) atomicAdd(out, s);
}

__global__ void update_cur_kernel(float* __restrict__ cur,
                                  const float* __restrict__ ver,
                                  const float* __restrict__ nec,
                                  const int* __restrict__ done, long long n) {
  if (*done) return;  // uniform across grid
  for (long long i = blockIdx.x * (long long)blockDim.x + threadIdx.x; i < n;
       i += (long long)gridDim.x * blockDim.x)
    cur[i] += ver[i] * nec[i];
}

__global__ void update_done_kernel(const float* __restrict__ sum, float inv_n,
                                   int* __restrict__ done) {
  if (sum[0] * inv_n > 0.9f) *done = 1;
}

// ---------------------------------------------------------------------------
extern "C" void kernel_launch(void* const* d_in, const int* in_sizes, int n_in,
                              void* d_out, int out_size, void* d_ws, size_t ws_size,
                              hipStream_t stream) {
  const float* x         = (const float*)d_in[0];
  const float* world_emb = (const float*)d_in[1];
  const float* acc       = (const float*)d_in[2];
  const float* mixer_W   = (const float*)d_in[3];
  const float* mixer_b   = (const float*)d_in[4];
  const float* pWq = (const float*)d_in[5];
  const float* pWk = (const float*)d_in[6];
  const float* pWv = (const float*)d_in[7];
  const float* pWp = (const float*)d_in[8];
  const float* p_mw = (const float*)d_in[9];
  const float* nWq = (const float*)d_in[10];
  const float* nWk = (const float*)d_in[11];
  const float* nWv = (const float*)d_in[12];
  const float* nWp = (const float*)d_in[13];
  const float* n_mw = (const float*)d_in[14];
  const float* vW1 = (const float*)d_in[15];
  const float* vb1 = (const float*)d_in[16];
  const float* vW2 = (const float*)d_in[17];
  const float* vb2 = (const float*)d_in[18];
  float* out = (float*)d_out;   // cur lives directly in d_out

  // workspace layout (floats)
  float* ws = (float*)d_ws;
  size_t off = 0;
  float* Weff = ws + off; off += (size_t)C_DIM * C_DIM;
  float* beff = ws + off; off += C_DIM;
  float* qb   = ws + off; off += NEL;
  float* kb   = ws + off; off += NEL;
  float* vb   = ws + off; off += NEL;
  float* ab   = ws + off; off += NEL;   // raw attention output
  float* poss = ws + off; off += NEL;
  float* nec  = ws + off; off += NEL;
  float* hid  = ws + off; off += NEL;
  float* ver  = ws + off; off += NEL;
  float* sums = ws + off; off += 8;
  int*   done = (int*)(ws + off);

  hipMemsetAsync(sums, 0, 8 * sizeof(float) + sizeof(int), stream);

  // Kripke mixer (algebraically folded into one GEMM)
  prep_weff_kernel<<<(C_DIM * C_DIM + 255) / 256, 256, 0, stream>>>(mixer_W, acc, Weff);
  prep_beff_kernel<<<3, 256, 0, stream>>>(mixer_W, mixer_b, acc, world_emb, beff);

  const dim3 ggrid(M_TOK / 64, C_DIM / 32);   // (128, 24)
  const dim3 agrid(B_SZ * H_NUM, T_SEQ / 16); // (64, 64)

  linear_wmma_kernel<<<ggrid, 256, 0, stream>>>(x, Weff, beff, out, M_TOK, C_DIM, C_DIM, 0);

  for (int step = 0; step < 5; ++step) {
    // possibility attention
    linear_wmma_kernel<<<ggrid, 256, 0, stream>>>(out, pWq, nullptr, qb, M_TOK, C_DIM, C_DIM, 0);
    linear_wmma_kernel<<<ggrid, 256, 0, stream>>>(out, pWk, nullptr, kb, M_TOK, C_DIM, C_DIM, 0);
    linear_wmma_kernel<<<ggrid, 256, 0, stream>>>(out, pWv, nullptr, vb, M_TOK, C_DIM, C_DIM, 0);
    attn_wmma_kernel<<<agrid, 128, 0, stream>>>(qb, kb, vb, acc, p_mw, 0, ab);
    linear_wmma_kernel<<<ggrid, 256, 0, stream>>>(ab, pWp, nullptr, poss, M_TOK, C_DIM, C_DIM, 0);

    // necessity attention
    linear_wmma_kernel<<<ggrid, 256, 0, stream>>>(poss, nWq, nullptr, qb, M_TOK, C_DIM, C_DIM, 0);
    linear_wmma_kernel<<<ggrid, 256, 0, stream>>>(poss, nWk, nullptr, kb, M_TOK, C_DIM, C_DIM, 0);
    linear_wmma_kernel<<<ggrid, 256, 0, stream>>>(poss, nWv, nullptr, vb, M_TOK, C_DIM, C_DIM, 0);
    attn_wmma_kernel<<<agrid, 128, 0, stream>>>(qb, kb, vb, acc, n_mw, 1, ab);
    linear_wmma_kernel<<<ggrid, 256, 0, stream>>>(ab, nWp, nullptr, nec, M_TOK, C_DIM, C_DIM, 0);

    // verifier MLP
    linear_wmma_kernel<<<ggrid, 256, 0, stream>>>(nec, vW1, vb1, hid, M_TOK, C_DIM, C_DIM, 1);
    linear_wmma_kernel<<<ggrid, 256, 0, stream>>>(hid, vW2, vb2, ver, M_TOK, C_DIM, C_DIM, 2);

    // early-exit bookkeeping (reference order: update cur with OLD done, then done)
    reduce_sum_kernel<<<512, 256, 0, stream>>>(ver, (long long)NEL, &sums[step]);
    update_cur_kernel<<<2048, 256, 0, stream>>>(out, ver, nec, done, (long long)NEL);
    update_done_kernel<<<1, 1, 0, stream>>>(&sums[step], 1.0f / (float)NEL, done);
  }
}
